// RNN_70480413327462
// MI455X (gfx1250) — matrix operand order repaired and checked by reading, not compiled, not run on hardware
//
#include <hip/hip_runtime.h>

typedef __attribute__((ext_vector_type(16))) _Float16 v16h;
typedef __attribute__((ext_vector_type(8)))  float    v8f;

namespace {
constexpr int HH   = 32;    // hidden size
constexpr int EE   = 34;    // embed dim (32 hidden + 2 coords)
constexpr int EPAD = 36;    // padded row stride for Q/K/V + LDS tiles
constexpr int PP   = 4096;  // patches (64x64)

// ---- WMMA fragment loaders (contiguous k order, same in A and B, wave32) ----

// A fragment: 16x32 f16, element source is f32 row-major [16, ld].
// lane 0-15 : M=lane,    K = 0..15
// lane 16-31: M=lane-16, K = 16..31
__device__ __forceinline__ v16h load_a_f32(const float* __restrict__ base, int ld) {
  const int lane = threadIdx.x & 31;
  const int m  = lane & 15;
  const int k0 = (lane & 16) ? 16 : 0;
  const float* p = base + m * ld;
  v16h a;
#pragma unroll
  for (int e = 0; e < 8; ++e) a[e]     = (_Float16)p[k0 + e];
#pragma unroll
  for (int e = 0; e < 8; ++e) a[8 + e] = (_Float16)p[k0 + 8 + e];
  return a;
}

// B fragment: 32x16 f16 = transpose of W rows [outdim, indim] (computes X @ W^T).
// lane 0-15: N=lane, K=0..15 ; lane 16-31: N=lane-16, K=16..31 (2 K per VGPR).
__device__ __forceinline__ v16h load_b_wt(const float* __restrict__ W, int indim,
                                          int outdim, int colbase) {
  const int lane = threadIdx.x & 31;
  const int col  = colbase + (lane & 15);
  const int kb   = (lane & 16) ? 16 : 0;
  v16h b;
  if (col < outdim) {
    const float* wp = W + col * indim + kb;
#pragma unroll
    for (int e = 0; e < 16; ++e) b[e] = (_Float16)wp[e];
  } else {
#pragma unroll
    for (int e = 0; e < 16; ++e) b[e] = (_Float16)0.f;
  }
  return b;
}

// 9-neighbor flat index with the reference's boundary shift.
__device__ __forceinline__ int nbr_idx(int p, int l) {
  const int r = p >> 6, c = p & 63;
  int nr = r + (l / 3 - 1);
  int nc = c + (l % 3 - 1);
  if (r == 0)  nr += 1;
  if (r == 63) nr -= 1;
  if (c == 0)  nc += 1;
  if (c == 63) nc -= 1;
  return (nr << 6) + nc;
}
} // namespace

__global__ void zero_f32(float* __restrict__ p, int n) {
  const int i = blockIdx.x * blockDim.x + threadIdx.x;
  if (i < n) p[i] = 0.f;
}

// -------- Phase 1: hnew = tanh(x W_x2h^T + h W_h2h^T + b) ; Q/K/V = qin W_{q,k,v}^T + b
// One wave per 16-row tile, 8 waves (128 rows) per block.
__global__ __launch_bounds__(256) void rnn_phase1(
    const float* __restrict__ inp, const float* __restrict__ hIn,
    const float* __restrict__ Wx2h, const float* __restrict__ bx2h,
    const float* __restrict__ Wh2h, const float* __restrict__ bh2h,
    const float* __restrict__ Win,  const float* __restrict__ bin,
    float* __restrict__ hA, float* __restrict__ Qb,
    float* __restrict__ Kb, float* __restrict__ Vb, int N) {
  __shared__ float s_h[8][16 * EPAD];
  const int wave = threadIdx.x >> 5;
  const int lane = threadIdx.x & 31;
  const int rowbase = (blockIdx.x * 8 + wave) * 16;
  float* ht = s_h[wave];
  const int nlo  = lane & 15;
  const int moff = (lane & 16) ? 8 : 0;
  (void)N;

  // ---- hnew (two 16-col tiles, K=32 each, two WMMAs chained per tile) ----
  v16h a_in = load_a_f32(inp + (size_t)rowbase * HH, HH);
  v16h a_h  = load_a_f32(hIn + (size_t)rowbase * HH, HH);
#pragma unroll
  for (int ct = 0; ct < 2; ++ct) {
    v16h b1 = load_b_wt(Wx2h, HH, HH, ct * 16);
    v16h b2 = load_b_wt(Wh2h, HH, HH, ct * 16);
    v8f c = {};
    c = __builtin_amdgcn_wmma_f32_16x16x32_f16(false, a_in, false, b1, (short)0, c, false, false);
    c = __builtin_amdgcn_wmma_f32_16x16x32_f16(false, a_h,  false, b2, (short)0, c, false, false);
    const int ncol = ct * 16 + nlo;
    const float bias = bx2h[ncol] + bh2h[ncol];
#pragma unroll
    for (int r = 0; r < 8; ++r) {
      const int m = r + moff;
      const float v = tanhf(c[r] + bias);
      ht[m * EPAD + ncol] = v;
      hA[(size_t)(rowbase + m) * HH + ncol] = v;
    }
  }
  __syncthreads();

  // ---- Q/K/V: K=32 WMMA over hnew; coord cols (K=32,33) folded as rank-2 VALU term ----
  v16h a_q = load_a_f32(ht, EPAD);
#pragma unroll
  for (int mat = 0; mat < 3; ++mat) {
    const float* Wm = Win + mat * EE * EE;   // W_in rows [mat*34 .. mat*34+33]
    float* outb = (mat == 0) ? Qb : (mat == 1) ? Kb : Vb;
#pragma unroll
    for (int ct = 0; ct < 3; ++ct) {
      v16h b = load_b_wt(Wm, EE, EE, ct * 16);
      v8f c = {};
      c = __builtin_amdgcn_wmma_f32_16x16x32_f16(false, a_q, false, b, (short)0, c, false, false);
      const int ncol = ct * 16 + nlo;
      if (ncol < EE) {
        const float w32 = Wm[ncol * EE + 32];
        const float w33 = Wm[ncol * EE + 33];
        const float bias = bin[mat * EE + ncol];
#pragma unroll
        for (int r = 0; r < 8; ++r) {
          const int m = r + moff;
          const int n = rowbase + m;
          const int p = n & (PP - 1);
          const float c0 = (float)(p >> 6) * (1.0f / 64.0f);
          const float c1 = (float)(p & 63) * (1.0f / 64.0f);
          outb[(size_t)n * EPAD + ncol] = c[r] + bias + c0 * w32 + c1 * w33;
        }
      }
    }
  }
}

// -------- Phase 2: center-query 9-neighbor attention + out/fcsa/fc2 + LayerNorm
__global__ __launch_bounds__(256) void rnn_phase2(
    const float* __restrict__ hA,
    const float* __restrict__ Qb, const float* __restrict__ Kb, const float* __restrict__ Vb,
    const float* __restrict__ Wout, const float* __restrict__ bout,
    const float* __restrict__ Wfcsa, const float* __restrict__ bfcsa,
    const float* __restrict__ Wfc2, const float* __restrict__ bfc2,
    const float* __restrict__ lng, const float* __restrict__ lnb,
    float* __restrict__ hOut, float* __restrict__ pred, float* __restrict__ prevPred,
    int N) {
  __shared__ float sA[8][16 * EPAD];
  __shared__ float sB[8][16 * EPAD];
  const int wave = threadIdx.x >> 5;
  const int lane = threadIdx.x & 31;
  const int rowbase = (blockIdx.x * 8 + wave) * 16;
  float* bufA = sA[wave];
  float* bufB = sB[wave];
  const int nlo  = lane & 15;
  const int moff = (lane & 16) ? 8 : 0;
  (void)N;

  // ---- attention (one row per lane 0..15): gather neighbor K/V, softmax, o -> bufA ----
  if (lane < 16) {
    const int n = rowbase + lane;
    const int b = n >> 12;              // row / P
    const int p = n & (PP - 1);
    const int base = b << 12;

    float q[EE];
    {
      const float* qp = Qb + (size_t)(base + nbr_idx(p, 4)) * EPAD;  // center query
#pragma unroll
      for (int e = 0; e < EE; ++e) q[e] = qp[e];
    }
    float s[9];
    float smax = -1e30f;
#pragma unroll
    for (int l = 0; l < 9; ++l) {
      const float* kp = Kb + (size_t)(base + nbr_idx(p, l)) * EPAD;
      float acc = 0.f;
#pragma unroll
      for (int e = 0; e < EE; ++e) acc += q[e] * kp[e];
      s[l] = acc * 0.1714985851425088f;   // 1/sqrt(34)
      smax = fmaxf(smax, s[l]);
    }
    float sum = 0.f;
#pragma unroll
    for (int l = 0; l < 9; ++l) { s[l] = expf(s[l] - smax); sum += s[l]; }
    const float inv = 1.f / sum;
    float o[EE];
#pragma unroll
    for (int e = 0; e < EE; ++e) o[e] = 0.f;
#pragma unroll
    for (int l = 0; l < 9; ++l) {
      const float w = s[l] * inv;
      const float* vp = Vb + (size_t)(base + nbr_idx(p, l)) * EPAD;
#pragma unroll
      for (int e = 0; e < EE; ++e) o[e] += w * vp[e];
    }
#pragma unroll
    for (int e = 0; e < EE; ++e) bufA[lane * EPAD + e] = o[e];
  }
  __syncthreads();

  // ---- ao = o @ Wout^T + bout  (K=32 WMMA + K=32..33 remainder) -> bufB ----
  {
    v16h a_o = load_a_f32(bufA, EPAD);
#pragma unroll
    for (int ct = 0; ct < 3; ++ct) {
      v16h b = load_b_wt(Wout, EE, EE, ct * 16);
      v8f c = {};
      c = __builtin_amdgcn_wmma_f32_16x16x32_f16(false, a_o, false, b, (short)0, c, false, false);
      const int ncol = ct * 16 + nlo;
      if (ncol < EE) {
        const float w32 = Wout[ncol * EE + 32];
        const float w33 = Wout[ncol * EE + 33];
        const float bias = bout[ncol];
#pragma unroll
        for (int r = 0; r < 8; ++r) {
          const int m = r + moff;
          bufB[m * EPAD + ncol] =
              c[r] + bias + bufA[m * EPAD + 32] * w32 + bufA[m * EPAD + 33] * w33;
        }
      }
    }
  }
  __syncthreads();

  // ---- h2 = hnew + ao @ Wfcsa^T + bfcsa -> bufA + hOut ----
  {
    v16h a_ao = load_a_f32(bufB, EPAD);
#pragma unroll
    for (int ct = 0; ct < 2; ++ct) {
      v16h b = load_b_wt(Wfcsa, EE, HH, ct * 16);
      v8f c = {};
      c = __builtin_amdgcn_wmma_f32_16x16x32_f16(false, a_ao, false, b, (short)0, c, false, false);
      const int ncol = ct * 16 + nlo;
      const float w32 = Wfcsa[ncol * EE + 32];
      const float w33 = Wfcsa[ncol * EE + 33];
      const float bias = bfcsa[ncol];
#pragma unroll
      for (int r = 0; r < 8; ++r) {
        const int m = r + moff;
        const float v = hA[(size_t)(rowbase + m) * HH + ncol] + c[r] + bias
                      + bufB[m * EPAD + 32] * w32 + bufB[m * EPAD + 33] * w33;
        bufA[m * EPAD + ncol] = v;
        hOut[(size_t)(rowbase + m) * HH + ncol] = v;
      }
    }
  }
  __syncthreads();

  // ---- t = h2 @ Wfc2^T + bfc2 -> bufB ----
  {
    v16h a_h2 = load_a_f32(bufA, EPAD);
#pragma unroll
    for (int ct = 0; ct < 2; ++ct) {
      v16h b = load_b_wt(Wfc2, HH, HH, ct * 16);
      v8f c = {};
      c = __builtin_amdgcn_wmma_f32_16x16x32_f16(false, a_h2, false, b, (short)0, c, false, false);
      const int ncol = ct * 16 + nlo;
      const float bias = bfc2[ncol];
#pragma unroll
      for (int r = 0; r < 8; ++r) bufB[(r + moff) * EPAD + ncol] = c[r] + bias;
    }
  }
  __syncthreads();

  // ---- LayerNorm per row, write pred + feedback buffer ----
  if (lane < 16) {
    const int n = rowbase + lane;
    const float* tr = bufB + lane * EPAD;
    float mu = 0.f;
#pragma unroll
    for (int j = 0; j < HH; ++j) mu += tr[j];
    mu *= (1.0f / HH);
    float var = 0.f;
#pragma unroll
    for (int j = 0; j < HH; ++j) { const float d = tr[j] - mu; var += d * d; }
    var *= (1.0f / HH);
    const float rinv = rsqrtf(var + 1e-5f);
#pragma unroll
    for (int j = 0; j < HH; ++j) {
      const float pv = (tr[j] - mu) * rinv * lng[j] + lnb[j];
      pred[(size_t)n * HH + j] = pv;
      prevPred[(size_t)n * HH + j] = pv;
    }
  }
}

extern "C" void kernel_launch(void* const* d_in, const int* in_sizes, int n_in,
                              void* d_out, int out_size, void* d_ws, size_t ws_size,
                              hipStream_t stream) {
  const float* x     = (const float*)d_in[0];
  const float* Wx2h  = (const float*)d_in[1];
  const float* bx2h  = (const float*)d_in[2];
  const float* Wh2h  = (const float*)d_in[3];
  const float* bh2h  = (const float*)d_in[4];
  const float* Wfc2  = (const float*)d_in[5];
  const float* bfc2  = (const float*)d_in[6];
  const float* lng   = (const float*)d_in[7];
  const float* lnb   = (const float*)d_in[8];
  const float* Wfcsa = (const float*)d_in[9];
  const float* bfcsa = (const float*)d_in[10];
  const float* Win   = (const float*)d_in[11];
  const float* bin   = (const float*)d_in[12];
  const float* Wout  = (const float*)d_in[13];
  const float* bout  = (const float*)d_in[14];

  const int N = in_sizes[0] / (10 * HH);     // B * 4096 rows

  float* ws       = (float*)d_ws;
  float* hState   = ws;                                   // [N,32]
  float* hA       = hState   + (size_t)N * HH;            // [N,32]
  float* Qb       = hA       + (size_t)N * HH;            // [N,36]
  float* Kb       = Qb       + (size_t)N * EPAD;          // [N,36]
  float* Vb       = Kb       + (size_t)N * EPAD;          // [N,36]
  float* prevPred = Vb       + (size_t)N * EPAD;          // [N,32]
  float* out = (float*)d_out;

  const int blocks = N / 128;                // 8 waves x 16 rows per block
  zero_f32<<<(N * HH + 255) / 256, 256, 0, stream>>>(hState, N * HH);

  for (int t = 0; t < 59; ++t) {
    const float* inp = (t < 10) ? (x + (size_t)t * N * HH) : prevPred;
    rnn_phase1<<<blocks, 256, 0, stream>>>(inp, hState, Wx2h, bx2h, Wh2h, bh2h,
                                           Win, bin, hA, Qb, Kb, Vb, N);
    rnn_phase2<<<blocks, 256, 0, stream>>>(hA, Qb, Kb, Vb, Wout, bout, Wfcsa, bfcsa,
                                           Wfc2, bfc2, lng, lnb,
                                           hState, out + (size_t)t * N * HH, prevPred, N);
  }
  (void)n_in; (void)out_size; (void)ws_size;
}
